// StyleGenerator_3917010174160
// MI455X (gfx1250) — compile-verified, matrix operand order checked by reading, not run on hardware
//
#include <hip/hip_runtime.h>
#include <hip/hip_bf16.h>
#include <math.h>

// ---------------------------------------------------------------------------
// StyleGAN2 generator forward for MI455X (gfx1250).
//  - Convolutions as implicit-GEMM with v_wmma_f32_16x16x32_bf16 (wave32).
//  - Wave computes 32 out-channels x 16 pixels: two WMMA accumulator chains
//    sharing one B fragment (halves act traffic, hides WMMA latency).
//  - Tap-outer / channel-chunk-inner loop: no divisions or bounds checks in
//    the hot loop; activations bf16 NHWC so B fragments are b128 loads.
//  - Upsample (lhs_dilation=2) conv via parity-mapped zero-insertion
//    addressing, then 4x4 FIR blur + bias/lrelu.
// Assumed d_in ordering (setup_inputs dict order, nested insertion order):
//   0 latent_code, 1 const,
//   2..5   conv1    {mod_w, mod_b, w, act_b}
//   6..9   to_rgb1  {mod_w, mod_b, w, bias}
//   10+4j  convs[j] {mod_w, mod_b, w, act_b}, j=0..11
//   58+4j  to_rgbs[j] {mod_w, mod_b, w, bias}, j=0..5
// ---------------------------------------------------------------------------

#define B_SZ 4
#define LAT  512

typedef __bf16 bf16_t;
typedef bf16_t   v16bf __attribute__((ext_vector_type(16)));
typedef float    v8f   __attribute__((ext_vector_type(8)));
typedef unsigned v4u   __attribute__((ext_vector_type(4)));

union Frag16 { v16bf v; v4u q[2]; };
union Acc8   { v8f v; float f[8]; };

__device__ __forceinline__ unsigned short f2bf(float f) {
  union { float f; unsigned u; } c; c.f = f;
  unsigned u = c.u + 0x7FFFu + ((c.u >> 16) & 1u);   // round-to-nearest-even
  return (unsigned short)(u >> 16);
}
__device__ __forceinline__ float bf2f(unsigned short h) {
  union { unsigned u; float f; } c; c.u = ((unsigned)h) << 16;
  return c.f;
}

// --------------------------- style GEMV: s = z @ (W^T/sqrt(512)) + b --------
__global__ void style_kernel(const float* __restrict__ latent,
                             const float* __restrict__ mod_w,
                             const float* __restrict__ mod_b,
                             float* __restrict__ s, int inC) {
  int tid = blockIdx.x * blockDim.x + threadIdx.x;
  if (tid >= B_SZ * inC) return;
  int b = tid / inC, i = tid - b * inC;
  const float* lr = latent + (size_t)b * LAT;
  const float* wr = mod_w + (size_t)i * LAT;
  float acc = 0.f;
  for (int d = 0; d < LAT; ++d) acc += lr[d] * wr[d];
  s[tid] = acc * rsqrtf((float)LAT) + mod_b[i];
}

// ------------- per-batch weight modulate (+demod) -> packed bf16 ------------
// wgt[(b*outC+o)*9*inC + tap*inC + i], optional spatial flip (upsample path)
__global__ void modw_kernel(const float* __restrict__ w,
                            const float* __restrict__ s,
                            unsigned short* __restrict__ wgt,
                            int inC, int outC, int demod, int flip) {
  __shared__ float red[256];
  __shared__ float dem_s;
  const int bo = blockIdx.x;
  const int b = bo / outC, o = bo - b * outC;
  const int K = inC * 9;
  const float scale = rsqrtf((float)(inC * 9));
  const float* wo = w + (size_t)o * inC * 9;
  const float* sb = s + (size_t)b * inC;

  float local = 0.f;
  for (int idx = threadIdx.x; idx < K; idx += 256) {
    int i = idx / 9, t = idx - 9 * i;
    float v = scale * wo[(size_t)i * 9 + t] * sb[i];
    local += v * v;
  }
  red[threadIdx.x] = local;
  __syncthreads();
  for (int st = 128; st > 0; st >>= 1) {
    if (threadIdx.x < st) red[threadIdx.x] += red[threadIdx.x + st];
    __syncthreads();
  }
  if (threadIdx.x == 0) dem_s = demod ? rsqrtf(red[0] + 1e-8f) : 1.f;
  __syncthreads();
  const float dm = dem_s;

  unsigned short* wrow = wgt + (size_t)bo * K;
  for (int idx = threadIdx.x; idx < K; idx += 256) {
    int i = idx / 9, t = idx - 9 * i;
    float v = scale * wo[(size_t)i * 9 + t] * sb[i] * dm;
    int td = flip ? (8 - t) : t;
    wrow[(size_t)td * inC + i] = f2bf(v);
  }
}

// ---------------- seed: const [512,4,4] NCHW f32 -> [B,4,4,512] NHWC bf16 ---
__global__ void seed_kernel(const float* __restrict__ cst,
                            unsigned short* __restrict__ act) {
  int tid = blockIdx.x * blockDim.x + threadIdx.x;
  if (tid >= B_SZ * 16 * 512) return;
  int c = tid % 512, r = tid / 512;
  int p = r % 16, b = r / 16;
  act[((size_t)b * 16 + p) * 512 + c] = f2bf(cst[(size_t)c * 16 + p]);
}

// --------------------- implicit-GEMM 3x3 conv, bf16 WMMA --------------------
// Wave = 32 out-channels x 16 pixels (4x4 tile).  Tap-outer (unrolled),
// channel-chunk inner loop: 6 x b128 loads + 2 x WMMA per 32-channel chunk.
// up=0: pad-1 conv.  up=1: lhs_dilation=2 pad-2 conv via parity addressing.
__global__ void __launch_bounds__(32)
conv3x3_wmma(const unsigned short* __restrict__ act,
             const unsigned short* __restrict__ wgt,
             const float* __restrict__ act_b,
             unsigned short* __restrict__ out,
             int inH, int inW, int outH, int outW,
             int inC, int outC, int up, int apply_act) {
  const int lane = threadIdx.x;
  const int n = lane & 15;
  const int hioff = (lane >> 4) ? 8 : 0;
  const int tilesX = (outW + 3) >> 2;
  const int ty = (blockIdx.x / tilesX) << 2;
  const int tx = (blockIdx.x % tilesX) << 2;
  const int py = ty + (n >> 2);
  const int px = tx + (n & 3);
  const int b = blockIdx.z;
  const int oc0 = blockIdx.y << 5;                 // 32 out-channels / wave
  const int Ktot = 9 * inC;
  const unsigned short* wrow0 =
      wgt + ((size_t)(b * outC + oc0 + n)) * Ktot + hioff;
  const unsigned short* wrow1 = wrow0 + (size_t)16 * Ktot;

  Acc8 acc0, acc1;
#pragma unroll
  for (int r = 0; r < 8; ++r) { acc0.f[r] = 0.f; acc1.f[r] = 0.f; }

#pragma unroll
  for (int t = 0; t < 9; ++t) {
    const int dy = t / 3;                          // compile-time constants
    const int dx = t - 3 * dy;
    int iy, ix;
    bool valid;
    if (!up) {
      iy = py + dy - 1; ix = px + dx - 1;
      valid = (iy >= 0) && (ix >= 0) && (iy < inH) && (ix < inW);
    } else {
      const int zy = py + dy - 2, zx = px + dx - 2;
      valid = (zy >= 0) && (zx >= 0) &&
              (zy <= 2 * inH - 2) && (zx <= 2 * inW - 2) &&
              ((zy & 1) == 0) && ((zx & 1) == 0);
      iy = zy >> 1; ix = zx >> 1;
    }
    const unsigned short* ap =
        act + (((size_t)b * inH + iy) * inW + ix) * inC + hioff;
    const unsigned short* wt0 = wrow0 + (size_t)t * inC;
    const unsigned short* wt1 = wrow1 + (size_t)t * inC;

    for (int ic = 0; ic < inC; ic += 32) {
      Frag16 a0, a1, bbf;
      a0.q[0] = *(const v4u*)(wt0 + ic);
      a0.q[1] = *(const v4u*)(wt0 + ic + 16);
      a1.q[0] = *(const v4u*)(wt1 + ic);
      a1.q[1] = *(const v4u*)(wt1 + ic + 16);
      if (valid) {
        bbf.q[0] = *(const v4u*)(ap + ic);
        bbf.q[1] = *(const v4u*)(ap + ic + 16);
      } else {
        v4u z = {0u, 0u, 0u, 0u};
        bbf.q[0] = z; bbf.q[1] = z;
      }
      acc0.v = __builtin_amdgcn_wmma_f32_16x16x32_bf16(
          false, a0.v, false, bbf.v, (short)0, acc0.v, false, false);
      acc1.v = __builtin_amdgcn_wmma_f32_16x16x32_bf16(
          false, a1.v, false, bbf.v, (short)0, acc1.v, false, false);
    }
  }

  if (px < outW && py < outH) {
    const size_t pixBase = (((size_t)b * outH + py) * outW + px) * outC;
#pragma unroll
    for (int g = 0; g < 2; ++g) {
      const Acc8& acc = g ? acc1 : acc0;
      const int mb = oc0 + g * 16 + hioff;
      unsigned pk[4];
#pragma unroll
      for (int r = 0; r < 8; ++r) {
        float v = acc.f[r];
        if (apply_act) {
          v += act_b[mb + r];
          v = (v > 0.f) ? v : 0.2f * v;
          v *= 1.4142135623730951f;
        }
        const unsigned short h = f2bf(v);
        if (r & 1) pk[r >> 1] |= ((unsigned)h) << 16;
        else       pk[r >> 1] = (unsigned)h;
      }
      v4u st = {pk[0], pk[1], pk[2], pk[3]};
      *(v4u*)(out + pixBase + mb) = st;
    }
  }
}

// --------- 4x4 FIR blur (x4) on the (2H+1)^2 raw conv + bias/lrelu ---------
__global__ void blur_act_kernel(const unsigned short* __restrict__ in,
                                unsigned short* __restrict__ out,
                                const float* __restrict__ act_b,
                                int H, int W, int C, int inH, int inW) {
  const float k1[4] = {1.f, 3.f, 3.f, 1.f};
  size_t tid = blockIdx.x * (size_t)blockDim.x + threadIdx.x;
  size_t total = (size_t)B_SZ * H * W * C;
  if (tid >= total) return;
  int c = (int)(tid % C); size_t r = tid / C;
  int x = (int)(r % W); r /= W;
  int y = (int)(r % H); int b = (int)(r / H);
  float s = 0.f;
  for (int u = 0; u < 4; ++u) {
    int iy = y + u - 1;
    if (iy < 0 || iy >= inH) continue;
    for (int v = 0; v < 4; ++v) {
      int ix = x + v - 1;
      if (ix < 0 || ix >= inW) continue;
      s += k1[u] * k1[v] * bf2f(in[(((size_t)b * inH + iy) * inW + ix) * C + c]);
    }
  }
  s *= (1.f / 16.f);               // blur2d * 4 = outer(k1,k1)/16
  s += act_b[c];
  s = (s > 0.f) ? s : 0.2f * s;
  s *= 1.4142135623730951f;
  out[(((size_t)b * H + y) * W + x) * C + c] = f2bf(s);
}

// ---- to_rgb: 1x1 modulated conv (no demod) + bias + upfirdn2d(skip,up=2) ---
__global__ void torgb_kernel(const unsigned short* __restrict__ act,
                             const float* __restrict__ w,
                             const float* __restrict__ bias,
                             const float* __restrict__ style,
                             const float* __restrict__ skipPrev,
                             float* __restrict__ dst,
                             int H, int W, int inC, int prevH, int prevW,
                             int hasSkip) {
  const float k1[4] = {1.f, 3.f, 3.f, 1.f};
  size_t tid = blockIdx.x * (size_t)blockDim.x + threadIdx.x;
  size_t total = (size_t)B_SZ * 3 * H * W;
  if (tid >= total) return;
  int x = (int)(tid % W); size_t r = tid / W;
  int y = (int)(r % H); r /= H;
  int c = (int)(r % 3); int b = (int)(r / 3);

  const float scale = rsqrtf((float)inC);
  const float* wr = w + (size_t)c * inC;
  const float* sr = style + (size_t)b * inC;
  const unsigned short* ar = act + (((size_t)b * H + y) * W + x) * inC;
  float acc = 0.f;
  for (int i = 0; i < inC; i += 8) {               // b128 activation loads
    v4u qa = *(const v4u*)(ar + i);
#pragma unroll
    for (int k = 0; k < 4; ++k) {
      unsigned u = qa[k];
      acc += bf2f((unsigned short)(u & 0xffffu)) * wr[i + 2 * k] * sr[i + 2 * k];
      acc += bf2f((unsigned short)(u >> 16)) * wr[i + 2 * k + 1] * sr[i + 2 * k + 1];
    }
  }
  acc = acc * scale + bias[c];

  if (hasSkip) {   // upfirdn2d(skip, blur*4, up=2, pad0=2, pad1=1)
    for (int u = 0; u < 4; ++u) {
      int zy = y + u - 2;
      if (zy < 0 || zy > 2 * prevH - 2 || (zy & 1)) continue;
      for (int v = 0; v < 4; ++v) {
        int zx = x + v - 2;
        if (zx < 0 || zx > 2 * prevW - 2 || (zx & 1)) continue;
        acc += k1[u] * k1[v] * (1.f / 16.f) *
               skipPrev[(((size_t)b * 3 + c) * prevH + (zy >> 1)) * prevW + (zx >> 1)];
      }
    }
  }
  dst[(((size_t)b * 3 + c) * H + y) * W + x] = acc;
}

// ---------------------------------------------------------------------------
extern "C" void kernel_launch(void* const* d_in, const int* in_sizes, int n_in,
                              void* d_out, int out_size, void* d_ws, size_t ws_size,
                              hipStream_t stream) {
  (void)in_sizes; (void)n_in; (void)out_size; (void)ws_size;
  const float* latent = (const float*)d_in[0];
  const float* cst    = (const float*)d_in[1];
  auto P = [&](int i) { return (const float*)d_in[i]; };

  // bump allocator over d_ws (~213 MB needed)
  char* wsp = (char*)d_ws;
  size_t off = 0;
  auto alloc = [&](size_t bytes) -> void* {
    void* p = wsp + off;
    off += (bytes + 255) & ~(size_t)255;
    return p;
  };
  float* styleBuf = (float*)alloc((size_t)B_SZ * 512 * sizeof(float));
  unsigned short* wgt = (unsigned short*)alloc((size_t)B_SZ * 512 * 9 * 512 * 2);
  const size_t actBytes = (size_t)B_SZ * 257 * 257 * 128 * 2;  // covers all maps
  unsigned short* actA = (unsigned short*)alloc(actBytes);
  unsigned short* actB = (unsigned short*)alloc(actBytes);
  unsigned short* actR = (unsigned short*)alloc(actBytes);
  float* S0 = (float*)alloc((size_t)B_SZ * 3 * 128 * 128 * sizeof(float));
  float* S1 = (float*)alloc((size_t)B_SZ * 3 * 128 * 128 * sizeof(float));

  auto cdiv = [](size_t a, size_t b) { return (a + b - 1) / b; };

  auto runStyle = [&](const float* mw, const float* mb, int inC) {
    size_t tot = (size_t)B_SZ * inC;
    style_kernel<<<dim3((unsigned)cdiv(tot, 256)), dim3(256), 0, stream>>>(
        latent, mw, mb, styleBuf, inC);
  };
  auto runModW = [&](const float* w, int inC, int outC, int demod, int flip) {
    modw_kernel<<<dim3(B_SZ * outC), dim3(256), 0, stream>>>(
        w, styleBuf, wgt, inC, outC, demod, flip);
  };
  auto runConv = [&](const unsigned short* in, unsigned short* outp,
                     const float* actb, int inH, int outH, int inC, int outC,
                     int up, int actflag) {
    unsigned tiles = (unsigned)(cdiv(outH, 4) * cdiv(outH, 4));
    conv3x3_wmma<<<dim3(tiles, outC / 32, B_SZ), dim3(32), 0, stream>>>(
        in, wgt, actb, outp, inH, inH, outH, outH, inC, outC, up, actflag);
  };
  auto runBlurAct = [&](const unsigned short* in, unsigned short* outp,
                        const float* actb, int H, int C) {
    size_t tot = (size_t)B_SZ * H * H * C;
    blur_act_kernel<<<dim3((unsigned)cdiv(tot, 256)), dim3(256), 0, stream>>>(
        in, outp, actb, H, H, C, H + 1, H + 1);
  };
  auto runToRGB = [&](const unsigned short* actp, const float* w,
                      const float* bias, const float* skipPrev, float* dstp,
                      int H, int inC, int prevH) {
    size_t tot = (size_t)B_SZ * 3 * H * H;
    torgb_kernel<<<dim3((unsigned)cdiv(tot, 256)), dim3(256), 0, stream>>>(
        actp, w, bias, styleBuf, skipPrev, dstp, H, H, inC, prevH, prevH,
        skipPrev != nullptr ? 1 : 0);
  };

  // seed const -> NHWC bf16
  {
    size_t tot = (size_t)B_SZ * 16 * 512;
    seed_kernel<<<dim3((unsigned)cdiv(tot, 256)), dim3(256), 0, stream>>>(cst, actA);
  }
  unsigned short* cur = actA;
  unsigned short* nxt = actB;

  // conv1 (styled, res 4, 512->512)
  runStyle(P(2), P(3), 512);
  runModW(P(4), 512, 512, 1, 0);
  runConv(cur, nxt, P(5), 4, 4, 512, 512, 0, 1);
  { unsigned short* t = cur; cur = nxt; nxt = t; }

  // to_rgb1 (res 4) -> S0
  runStyle(P(6), P(7), 512);
  runToRGB(cur, P(8), P(9), nullptr, S0, 4, 512, 0);
  float* skip = S0;
  float* skipAlt = S1;

  const int convIn [12] = {512,512,512,512,512,512,512,512,512,256,256,128};
  const int convOut[12] = {512,512,512,512,512,512,512,512,256,256,128,128};
  const int rgbIn  [6]  = {512,512,512,512,256,128};

  for (int j = 0; j < 6; ++j) {
    const int res = 8 << j;
    const int Hin = res >> 1;
    const int c0 = 2 * j, c1 = 2 * j + 1;

    // upsample styled conv: WMMA dilated conv -> raw (2H+1)^2 -> blur+act
    runStyle(P(10 + 4 * c0), P(10 + 4 * c0 + 1), convIn[c0]);
    runModW(P(10 + 4 * c0 + 2), convIn[c0], convOut[c0], 1, 1);
    runConv(cur, actR, nullptr, Hin, 2 * Hin + 1, convIn[c0], convOut[c0], 1, 0);
    runBlurAct(actR, nxt, P(10 + 4 * c0 + 3), res, convOut[c0]);
    { unsigned short* t = cur; cur = nxt; nxt = t; }

    // regular styled conv
    runStyle(P(10 + 4 * c1), P(10 + 4 * c1 + 1), convIn[c1]);
    runModW(P(10 + 4 * c1 + 2), convIn[c1], convOut[c1], 1, 0);
    runConv(cur, nxt, P(10 + 4 * c1 + 3), res, res, convIn[c1], convOut[c1], 0, 1);
    { unsigned short* t = cur; cur = nxt; nxt = t; }

    // to_rgb + upsampled skip
    runStyle(P(58 + 4 * j), P(58 + 4 * j + 1), rgbIn[j]);
    float* dst = (j == 5) ? (float*)d_out : skipAlt;
    runToRGB(cur, P(58 + 4 * j + 2), P(58 + 4 * j + 3), skip, dst, res,
             rgbIn[j], Hin);
    float* old = skip; skip = dst; skipAlt = old;
  }
}
